// NgramReranker_20289425507010
// MI455X (gfx1250) — compile-verified, matrix operand order checked by reading, not run on hardware
//
#include <hip/hip_runtime.h>
#include <hip/hip_bf16.h>

typedef __attribute__((ext_vector_type(16))) _Float16 v16h;
typedef __attribute__((ext_vector_type(8)))  float    v8f;

#define LSRC  64
#define LCAND 32
#define KCAND 16

__device__ __forceinline__ unsigned ballot32(int p) {
#if __has_builtin(__builtin_amdgcn_ballot_w32)
  return __builtin_amdgcn_ballot_w32(p != 0);
#else
  return (unsigned)__ballot(p);
#endif
}

__global__ __launch_bounds__(512) void ngram_rerank_kernel(
    const int* __restrict__ cand, const int* __restrict__ src,
    const int* __restrict__ padp, float* __restrict__ outTok,
    float* __restrict__ outLen, float* __restrict__ outScore)
{
  __shared__ int   sSrc[LSRC];
  __shared__ int   sCand[KCAND * LCAND];
  __shared__ float sScore[KCAND];

  const int b    = blockIdx.x;
  const int tid  = threadIdx.x;
  const int lane = tid & 31;
  const int wave = tid >> 5;
  const int pad  = padp[0];

  // ---- CDNA5 async global->LDS staging (ASYNCcnt path) ----
  {
    unsigned ldsC = (unsigned)(uintptr_t)(&sCand[tid]);          // low 32 bits = LDS byte offset
    unsigned offC = (unsigned)tid * 4u;
    unsigned long long baseC =
        (unsigned long long)(uintptr_t)(cand + (size_t)b * (KCAND * LCAND));
    asm volatile("global_load_async_to_lds_b32 %0, %1, %2"
                 :: "v"(ldsC), "v"(offC), "s"(baseC) : "memory");
    if (tid < LSRC) {   // waves 0-1 fully active: EXEC stays full per wave
      unsigned ldsS = (unsigned)(uintptr_t)(&sSrc[tid]);
      unsigned offS = (unsigned)tid * 4u;
      unsigned long long baseS =
          (unsigned long long)(uintptr_t)(src + (size_t)b * LSRC);
      asm volatile("global_load_async_to_lds_b32 %0, %1, %2"
                   :: "v"(ldsS), "v"(offS), "s"(baseS) : "memory");
    }
    asm volatile("s_wait_asynccnt 0" ::: "memory");
  }
  __syncthreads();

  // ---- wave `wave` scores candidate `wave`: lane j holds candidate token j ----
  const int c  = sCand[wave * LCAND + lane];
  const int cn = (c != pad);

  // dot = sum over vocab of count_src[v]*count_cand[v]
  //     = # ordered pairs (i,j): src_i == cand_j, both non-pad
  unsigned dot = 0;
  #pragma unroll 8
  for (int i = 0; i < LSRC; ++i) {
    int s = sSrc[i];                                    // uniform scalar broadcast
    dot += __popc(ballot32(cn && (c == s)));
  }

  // ||cand||^2 = # ordered equal pairs among candidate tokens
  unsigned nc2 = 0;
  #pragma unroll 8
  for (int j = 0; j < LCAND; ++j) {
    int cj = __shfl(c, j, 32);
    nc2 += __popc(ballot32(cn && (c == cj)));
  }

  // ||src||^2 (computed redundantly per wave; 2 tokens per lane)
  const int sa = sSrc[lane];
  const int sb = sSrc[lane + 32];
  const int san = (sa != pad), sbn = (sb != pad);
  unsigned ns2 = 0;
  #pragma unroll 8
  for (int i = 0; i < LSRC; ++i) {
    int s = sSrc[i];
    ns2 += __popc(ballot32(san && (sa == s)));
    ns2 += __popc(ballot32(sbn && (sb == s)));
  }

  const float denom = sqrtf((float)ns2) * sqrtf((float)nc2);
  const float score = 1.0f - (float)dot / denom;
  if (lane == 0) sScore[wave] = score;
  __syncthreads();

  // ---- stable descending argsort of 16 scores via one WMMA ----
  // A[m][k'] = 1 if candidate k' ranks strictly above m; rank[m] = (A x ones)[m][*]
  // A layout (16-bit 16x32): lanes 0-15 -> row m=lane, h0..7 = K 0..7, h8..15 = K16..23 (zero)
  //                          lanes16-31 -> row m=lane-16, h0..7 = K 8..15, h8..15 = K24..31 (zero)
  const int   m  = lane & 15;
  const int   kb = (lane >> 4) << 3;          // 0 or 8
  const float sm = sScore[m];
  v16h A;
  #pragma unroll
  for (int t = 0; t < 16; ++t) A[t] = (_Float16)0.0f;
  #pragma unroll
  for (int t = 0; t < 8; ++t) {
    int   kp = kb + t;
    float sk = sScore[kp];
    int above = (sk > sm) || ((sk == sm) && (kp < m));
    A[t] = (_Float16)(above ? 1.0f : 0.0f);
  }
  v16h Bones;
  #pragma unroll
  for (int t = 0; t < 16; ++t) Bones[t] = (_Float16)1.0f;
  v8f Cacc = {};
  v8f D = __builtin_amdgcn_wmma_f32_16x16x32_f16(
      /*neg_a=*/false, A, /*neg_b=*/false, Bones,
      /*c_mod=*/(short)0, Cacc, /*reuse_a=*/false, /*reuse_b=*/false);

  // D layout: VGPR v -> rank[M=v] on lanes 0-15, rank[M=8+v] on lanes 16-31.
  const int ks = wave & 7;
  float rv = D[0];
  #pragma unroll
  for (int t = 1; t < 8; ++t) rv = (ks == t) ? D[t] : rv;
  const int srcLane = (wave >= 8) ? 16 : 0;
  const float rf = __shfl(rv, srcLane, 32);
  const int rank = (int)(rf + 0.5f);

  // ---- outputs (all as float, concatenated: tokens | lengths | sorted scores) ----
  if (lane == 0) outScore[b * KCAND + rank] = score;
  if (rank == 0) {
    outTok[b * LCAND + lane] = (float)c;
    unsigned nz = __popc(ballot32(cn));
    if (lane == 0) outLen[b] = (float)nz;
  }
}

extern "C" void kernel_launch(void* const* d_in, const int* in_sizes, int n_in,
                              void* d_out, int out_size, void* d_ws, size_t ws_size,
                              hipStream_t stream) {
  (void)n_in; (void)out_size; (void)d_ws; (void)ws_size;
  const int* cand = (const int*)d_in[0];   // (B, K, 32) int32
  const int* src  = (const int*)d_in[1];   // (B, 64)    int32
  const int* pad  = (const int*)d_in[3];   // scalar pad_id
  const int  B    = in_sizes[1] / LSRC;

  float* out      = (float*)d_out;
  float* outTok   = out;                        // B*32
  float* outLen   = outTok + (size_t)B * LCAND; // B
  float* outScore = outLen + B;                 // B*16

  ngram_rerank_kernel<<<dim3(B), dim3(512), 0, stream>>>(
      cand, src, pad, outTok, outLen, outScore);
}
